// DFlashAttention_5488968204555
// MI455X (gfx1250) — matrix operand order repaired and checked by reading, not compiled, not run on hardware
//
#include <hip/hip_runtime.h>
#include <math.h>
#include <stddef.h>

// ---------- problem constants ----------
#define Bsz   4
#define QL    128
#define CTX   8192
#define KVL   (CTX + QL)        // 8320
#define HID   4096
#define NH    32
#define NKVH  8
#define HD    128               // head dim

typedef __attribute__((ext_vector_type(16))) __bf16 v16bf;
typedef __attribute__((ext_vector_type(8)))  float  v8f;
typedef __attribute__((ext_vector_type(4)))  int    i4vec;

#define WMMA_BF16(a, b, c) \
    __builtin_amdgcn_wmma_f32_16x16x32_bf16(false, (a), false, (b), (short)0, (c), false, false)

// ---- CDNA5 async global->LDS path (guarded; fallback = load + ds_store) ----
#if __has_builtin(__builtin_amdgcn_global_load_async_to_lds_b128) && \
    __has_builtin(__builtin_amdgcn_s_wait_asynccnt)
#define HAS_ASYNC_LDS 1
typedef __attribute__((address_space(1))) i4vec* gptr_i4;
typedef __attribute__((address_space(3))) i4vec* lptr_i4;
#define CP_WAIT(n) __builtin_amdgcn_s_wait_asynccnt(n)
#else
#define HAS_ASYNC_LDS 0
#define CP_WAIT(n)
#endif

// ---- scheduling directives: force loads | cvts | wmma pipeline shape ----
#if __has_builtin(__builtin_amdgcn_sched_group_barrier)
#define SCHED_GROUP(mask, size, id) __builtin_amdgcn_sched_group_barrier(mask, size, id)
#else
#define SCHED_GROUP(mask, size, id)
#endif

union Frag {
    v16bf  v;
    __bf16 e[16];
    uint4  q[2];
};

__device__ __forceinline__ void cvt8(Frag& f, int base, float4 a, float4 b) {
    f.e[base + 0] = (__bf16)a.x; f.e[base + 1] = (__bf16)a.y;
    f.e[base + 2] = (__bf16)a.z; f.e[base + 3] = (__bf16)a.w;
    f.e[base + 4] = (__bf16)b.x; f.e[base + 5] = (__bf16)b.y;
    f.e[base + 6] = (__bf16)b.z; f.e[base + 7] = (__bf16)b.w;
}

// copy 16 bytes global -> LDS at byte offset OFF, async when builtin exists
template <int OFF>
__device__ __forceinline__ void cp16(const __bf16* g, __bf16* l) {
#if HAS_ASYNC_LDS
    __builtin_amdgcn_global_load_async_to_lds_b128(
        (gptr_i4)((const char*)g + OFF), (lptr_i4)((char*)l + OFF), 0, 0);
#else
    *(uint4*)((char*)l + OFF) = *(const uint4*)((const char*)g + OFF);
#endif
}

// row pointer into (possibly concatenated) fp32 activation matrix
__device__ __forceinline__ const float* a_row(const float* A0, const float* A1,
                                              int m, int concat) {
    if (!concat) return A0 + (size_t)m * HID;
    int b = m / KVL, p = m % KVL;
    if (p < CTX) return A1 + ((size_t)b * CTX + p) * (size_t)HID;   // target_hidden
    return A0 + ((size_t)b * QL + (p - CTX)) * (size_t)HID;          // hidden_states
}

__device__ __forceinline__ void store_proj(__bf16* out, int m, int n, float v, int mode) {
    if (mode == 0) {            // K: [b][kvh][p][d]
        int b = m / KVL, p = m % KVL, kvh = n >> 7, d = n & 127;
        out[(((size_t)(b * NKVH + kvh)) * KVL + p) * HD + d] = (__bf16)v;
    } else if (mode == 1) {     // V transposed: [b][kvh][d][p]
        int b = m / KVL, p = m % KVL, kvh = n >> 7, d = n & 127;
        out[(((size_t)(b * NKVH + kvh)) * HD + d) * KVL + p] = (__bf16)v;
    } else {                    // Q: [b][h][q][d]
        int b = m >> 7, qq = m & 127, hh = n >> 7, d = n & 127;
        out[(((size_t)(b * NH + hh)) * QL + qq) * HD + d] = (__bf16)v;
    }
}

// ---------------------------------------------------------------------------
// Kernel 1: fp32 activations x fp32 weight -> bf16 output via bf16 WMMA.
// One wave -> 16x64 tile. Pipeline shape pinned with sched_group_barrier:
// 20 VMEM reads | 48 VALU cvt | 4 WMMA per k-step.
// ---------------------------------------------------------------------------
__global__ __launch_bounds__(256, 2)
void proj_kernel(const float* __restrict__ A0, const float* __restrict__ A1,
                 const float* __restrict__ W, __bf16* __restrict__ out,
                 int M, int N, int concat, int mode) {
    int lane = threadIdx.x & 31;
    int wid  = (blockIdx.x * blockDim.x + threadIdx.x) >> 5;
    int ngr  = N >> 6;
    int mt   = wid / ngr, ng = wid % ngr;
    if (mt * 16 >= M) return;
    int m0 = mt * 16, n0 = ng * 64;
    int mloc = lane & 15, h = lane >> 4;

    const float* arow = a_row(A0, A1, m0 + mloc, concat);
    const float* wrow0 = W + (size_t)(n0 + 0 * 16 + mloc) * HID + 16 * h;
    const float* wrow1 = W + (size_t)(n0 + 1 * 16 + mloc) * HID + 16 * h;
    const float* wrow2 = W + (size_t)(n0 + 2 * 16 + mloc) * HID + 16 * h;
    const float* wrow3 = W + (size_t)(n0 + 3 * 16 + mloc) * HID + 16 * h;

    v8f acc[4];
    for (int i = 0; i < 4; i++)
        for (int j = 0; j < 8; j++) acc[i][j] = 0.f;

#pragma unroll 2
    for (int kk = 0; kk < HID; kk += 32) {
        // ---- issue every load of this k-step first (20 x b128 in flight) ----
        const float* p1 = arow + kk + 16 * h;
        const float* p2 = arow + kk + 16 * h + 8;
        float4 a0 = *(const float4*)p1, a1 = *(const float4*)(p1 + 4);
        float4 a2 = *(const float4*)p2, a3 = *(const float4*)(p2 + 4);
        const float* wp[4] = { wrow0 + kk, wrow1 + kk, wrow2 + kk, wrow3 + kk };
        float4 wr[4][4];
#pragma unroll
        for (int nt = 0; nt < 4; nt++)
#pragma unroll
            for (int j = 0; j < 4; j++) wr[nt][j] = *(const float4*)(wp[nt] + 4 * j);

        // ---- convert + WMMA ----
        Frag a;
        cvt8(a, 0, a0, a1);
        cvt8(a, 8, a2, a3);
#pragma unroll
        for (int nt = 0; nt < 4; nt++) {
            Frag bf;
            cvt8(bf, 0, wr[nt][0], wr[nt][1]);
            cvt8(bf, 8, wr[nt][2], wr[nt][3]);
            acc[nt] = WMMA_BF16(a.v, bf.v, acc[nt]);
        }

        // pin the pipeline: loads first, then converts, then matrix ops
        SCHED_GROUP(0x020, 20, 0);   // VMEM reads
        SCHED_GROUP(0x002, 48, 0);   // VALU (cvt_pk etc.)
        SCHED_GROUP(0x008, 4, 0);    // WMMA
    }

#pragma unroll
    for (int nt = 0; nt < 4; nt++)
#pragma unroll
        for (int r = 0; r < 8; r++)
            store_proj(out, m0 + r + 8 * h, n0 + nt * 16 + mloc, acc[nt][r], mode);
}

// ---------------------------------------------------------------------------
// Kernel 2: in-place RMSNorm + RoPE on bf16 rows of 128. One wave per row.
// ---------------------------------------------------------------------------
__global__ __launch_bounds__(256)
void norm_rope_kernel(__bf16* __restrict__ buf, const float* __restrict__ w,
                      const float* __restrict__ cosb, const float* __restrict__ sinb,
                      int rows, int len, int pos_off) {
    int wid = (blockIdx.x * blockDim.x + threadIdx.x) >> 5;
    if (wid >= rows) return;
    int lane = threadIdx.x & 31;
    int d0   = lane * 4;
    __bf16* row = buf + (size_t)wid * HD;

    union { uint2 u; __bf16 e[4]; } up;
    up.u = *(const uint2*)(row + d0);
    float x[4];
#pragma unroll
    for (int i = 0; i < 4; i++) x[i] = (float)up.e[i];

    float ss = x[0] * x[0] + x[1] * x[1] + x[2] * x[2] + x[3] * x[3];
#pragma unroll
    for (int o = 1; o < 32; o <<= 1) ss += __shfl_xor(ss, o, 32);
    float inv = rsqrtf(ss * (1.f / 128.f) + 1e-6f);

    int pos = wid % len + pos_off;
    const float* cp = cosb + (size_t)pos * HD + d0;
    const float* sp = sinb + (size_t)pos * HD + d0;

    float y[4];
#pragma unroll
    for (int i = 0; i < 4; i++) y[i] = x[i] * inv * w[d0 + i];

    union { uint2 u; __bf16 e[4]; } op;
#pragma unroll
    for (int i = 0; i < 4; i++) {
        float part = __shfl_xor(y[i], 16, 32);          // value at d^64
        float rot  = (lane < 16) ? -part : part;        // rotate_half
        op.e[i] = (__bf16)(y[i] * cp[i] + rot * sp[i]);
    }
    *(uint2*)(row + d0) = op.u;
}

// ---------------------------------------------------------------------------
// Kernel 3: flash attention. 8 waves/block share one (b,h); K/V chunks are
// staged once per block into double-buffered LDS via async global->LDS,
// pipelined with s_wait_asynccnt + barriers. One wave per 16-row q tile.
// ---------------------------------------------------------------------------
#define NC (KVL / 32)           // 260 chunks of 32 kv positions

__global__ __launch_bounds__(256, 2)
void attn_kernel(const __bf16* __restrict__ Q, const __bf16* __restrict__ K,
                 const __bf16* __restrict__ Vt, __bf16* __restrict__ O) {
    __shared__ __align__(16) __bf16 ktile[2][32][128];   // 2 x 8 KB
    __shared__ __align__(16) __bf16 vtile[2][128][32];   // 2 x 8 KB
    __shared__ __align__(16) __bf16 pshare[8][16][32];   // 8 KB

    int tid = threadIdx.x;
    int wv = tid >> 5, lane = tid & 31;
    int gw = blockIdx.x * 8 + wv;                 // 0..1023
    int qt = gw & 7, h = (gw >> 3) & 31, b = gw >> 8;
    int kvh  = h >> 2;                            // GQA group of 4
    int q0   = qt * 16;
    int mloc = lane & 15, hh = lane >> 4;

    const __bf16* kbase = K  + ((size_t)(b * NKVH + kvh)) * KVL * HD;
    const __bf16* vbase = Vt + ((size_t)(b * NKVH + kvh)) * HD * KVL;

    // per-thread slices for the cooperative chunk copy (4 x b128 per thread)
    const __bf16* gk_t = kbase + tid * 16;                 // K chunk: 32x128 contiguous
    int vd = tid >> 1, vhalf = tid & 1;
    const __bf16* gv_t = vbase + (size_t)vd * KVL + vhalf * 16;

    // Q fragments for all 4 k-slices of D=128
    Frag qf[4];
    const __bf16* qrow = Q + (((size_t)(b * NH + h)) * QL + q0 + mloc) * HD;
#pragma unroll
    for (int s = 0; s < 4; s++) {
        qf[s].q[0] = *(const uint4*)(qrow + s * 32 + 16 * hh);
        qf[s].q[1] = *(const uint4*)(qrow + s * 32 + 16 * hh + 8);
    }

    v8f acc[8];
    for (int i = 0; i < 8; i++)
        for (int j = 0; j < 8; j++) acc[i][j] = 0.f;
    float mrun[8], lrun[8];
#pragma unroll
    for (int r = 0; r < 8; r++) { mrun[r] = -3.0e38f; lrun[r] = 0.f; }

    const float scale = 0.08838834764831845f;     // 1/sqrt(128)
    __bf16(*ps)[32] = pshare[wv];

    // prefetch chunk 0 into buffer 0
    {
        __bf16* lk = &ktile[0][0][0] + tid * 16;
        cp16<0>(gk_t, lk); cp16<16>(gk_t, lk);
        __bf16* lv = &vtile[0][vd][vhalf * 16];
        cp16<0>(gv_t, lv); cp16<16>(gv_t, lv);
    }

    for (int c = 0; c < NC; ++c) {
        int bufi = c & 1;
        __syncthreads();                          // everyone done with buf[bufi^1]
        if (c + 1 < NC) {                         // prefetch chunk c+1
            int off = (c + 1) * 32;               // elements along kv
            const __bf16* gk = gk_t + (size_t)off * HD;
            __bf16* lk = &ktile[bufi ^ 1][0][0] + tid * 16;
            cp16<0>(gk, lk); cp16<16>(gk, lk);
            const __bf16* gv = gv_t + off;
            __bf16* lv = &vtile[bufi ^ 1][vd][vhalf * 16];
            cp16<0>(gv, lv); cp16<16>(gv, lv);
            CP_WAIT(4);                           // chunk c's 4 async ops done
        } else {
            CP_WAIT(0);
        }
        __syncthreads();                          // chunk c visible to all waves

        __bf16(*kt)[128] = ktile[bufi];
        __bf16(*vt)[32]  = vtile[bufi];

        // ---- S = Q K^T for two 16-wide tiles (from LDS) ----
        v8f s01[2];
#pragma unroll
        for (int t = 0; t < 2; t++) {
            const __bf16* krow = &kt[t * 16 + mloc][0];
            v8f st;
            for (int j = 0; j < 8; j++) st[j] = 0.f;
#pragma unroll
            for (int s = 0; s < 4; s++) {
                Frag kf;
                kf.q[0] = *(const uint4*)(krow + s * 32 + 16 * hh);
                kf.q[1] = *(const uint4*)(krow + s * 32 + 16 * hh + 8);
                st = WMMA_BF16(qf[s].v, kf.v, st);
            }
            s01[t] = st;
        }

        // ---- online softmax over the 32 new columns ----
        float al[8];
#pragma unroll
        for (int r = 0; r < 8; r++) {
            float a0 = s01[0][r] * scale, a1 = s01[1][r] * scale;
            float mx = fmaxf(a0, a1);
#pragma unroll
            for (int o = 1; o < 16; o <<= 1) mx = fmaxf(mx, __shfl_xor(mx, o, 32));
            float mnew  = fmaxf(mrun[r], mx);
            float alpha = __expf(mrun[r] - mnew);
            mrun[r] = mnew;
            float p0 = __expf(a0 - mnew), p1 = __expf(a1 - mnew);
            float rs = p0 + p1;
#pragma unroll
            for (int o = 1; o < 16; o <<= 1) rs += __shfl_xor(rs, o, 32);
            lrun[r] = lrun[r] * alpha + rs;
            al[r]   = alpha;
            s01[0][r] = p0; s01[1][r] = p1;
        }
#pragma unroll
        for (int dt = 0; dt < 8; dt++)
#pragma unroll
            for (int r = 0; r < 8; r++) acc[dt][r] *= al[r];

        // ---- P: C-layout -> LDS [m][k] -> A-fragment ----
#pragma unroll
        for (int r = 0; r < 8; r++) {
            int rr = r + 8 * hh;
            ps[rr][mloc]      = (__bf16)s01[0][r];
            ps[rr][16 + mloc] = (__bf16)s01[1][r];
        }
        Frag pf;
        pf.q[0] = *(const uint4*)(&ps[mloc][16 * hh]);
        pf.q[1] = *(const uint4*)(&ps[mloc][16 * hh + 8]);

        // ---- O += P V (V from LDS) ----
#pragma unroll
        for (int dt = 0; dt < 8; dt++) {
            const __bf16* vrow = &vt[dt * 16 + mloc][0] + 16 * hh;
            Frag vf;
            vf.q[0] = *(const uint4*)(vrow);
            vf.q[1] = *(const uint4*)(vrow + 8);
            acc[dt] = WMMA_BF16(pf.v, vf.v, acc[dt]);
        }
    }

    // ---- finalize & store to Obuf [b][q][h*128+d] ----
#pragma unroll
    for (int r = 0; r < 8; r++) {
        float invl = 1.f / lrun[r];
        int qrow_o = q0 + r + 8 * hh;
#pragma unroll
        for (int dt = 0; dt < 8; dt++) {
            size_t idx = ((size_t)b * QL + qrow_o) * (size_t)(NH * HD)
                       + h * HD + dt * 16 + mloc;
            O[idx] = (__bf16)(acc[dt][r] * invl);
        }
    }
}

// ---------------------------------------------------------------------------
// Kernel 4: output projection, bf16 A x fp32 Wo -> fp32 d_out.
// ---------------------------------------------------------------------------
__global__ __launch_bounds__(256, 2)
void oproj_kernel(const __bf16* __restrict__ A, const float* __restrict__ W,
                  float* __restrict__ out, int M, int N) {
    int lane = threadIdx.x & 31;
    int wid  = (blockIdx.x * blockDim.x + threadIdx.x) >> 5;
    int ngr  = N >> 6;
    int mt   = wid / ngr, ng = wid % ngr;
    if (mt * 16 >= M) return;
    int m0 = mt * 16, n0 = ng * 64;
    int mloc = lane & 15, h = lane >> 4;

    const __bf16* arow = A + (size_t)(m0 + mloc) * HID;
    const float* wrow0 = W + (size_t)(n0 + 0 * 16 + mloc) * HID + 16 * h;
    const float* wrow1 = W + (size_t)(n0 + 1 * 16 + mloc) * HID + 16 * h;
    const float* wrow2 = W + (size_t)(n0 + 2 * 16 + mloc) * HID + 16 * h;
    const float* wrow3 = W + (size_t)(n0 + 3 * 16 + mloc) * HID + 16 * h;

    v8f acc[4];
    for (int i = 0; i < 4; i++)
        for (int j = 0; j < 8; j++) acc[i][j] = 0.f;

#pragma unroll 2
    for (int kk = 0; kk < HID; kk += 32) {
        Frag a;
        a.q[0] = *(const uint4*)(arow + kk + 16 * h);
        a.q[1] = *(const uint4*)(arow + kk + 16 * h + 8);
        const float* wp[4] = { wrow0 + kk, wrow1 + kk, wrow2 + kk, wrow3 + kk };
        float4 wr[4][4];
#pragma unroll
        for (int nt = 0; nt < 4; nt++)
#pragma unroll
            for (int j = 0; j < 4; j++) wr[nt][j] = *(const float4*)(wp[nt] + 4 * j);
#pragma unroll
        for (int nt = 0; nt < 4; nt++) {
            Frag bf;
            cvt8(bf, 0, wr[nt][0], wr[nt][1]);
            cvt8(bf, 8, wr[nt][2], wr[nt][3]);
            acc[nt] = WMMA_BF16(a.v, bf.v, acc[nt]);
        }

        SCHED_GROUP(0x020, 18, 0);   // VMEM reads (2 A + 16 W)
        SCHED_GROUP(0x002, 40, 0);   // VALU (cvt_pk etc.)
        SCHED_GROUP(0x008, 4, 0);    // WMMA
    }
#pragma unroll
    for (int nt = 0; nt < 4; nt++)
#pragma unroll
        for (int r = 0; r < 8; r++)
            out[(size_t)(m0 + r + 8 * h) * N + n0 + nt * 16 + mloc] = acc[nt][r];
}

// ---------------------------------------------------------------------------
extern "C" void kernel_launch(void* const* d_in, const int* in_sizes, int n_in,
                              void* d_out, int out_size, void* d_ws, size_t ws_size,
                              hipStream_t stream) {
    const float* hidden = (const float*)d_in[0];
    const float* target = (const float*)d_in[1];
    const float* cosb   = (const float*)d_in[2];   // (B, KVL, 128), identical per b
    const float* sinb   = (const float*)d_in[3];
    const float* Wq     = (const float*)d_in[4];
    const float* Wk     = (const float*)d_in[5];
    const float* Wv     = (const float*)d_in[6];
    const float* Wo     = (const float*)d_in[7];
    const float* qnw    = (const float*)d_in[8];
    const float* knw    = (const float*)d_in[9];
    (void)in_sizes; (void)n_in; (void)out_size; (void)ws_size;

    // workspace layout (bf16 elements)
    const size_t KV_ELTS = (size_t)Bsz * NKVH * KVL * HD;   // 34,078,720
    const size_t Q_ELTS  = (size_t)Bsz * NH * QL * HD;      //  2,097,152
    __bf16* Kbuf = (__bf16*)d_ws;
    __bf16* Vt   = Kbuf + KV_ELTS;
    __bf16* Qb   = Vt + KV_ELTS;
    __bf16* Ob   = Qb + Q_ELTS;

    const int MKV = Bsz * KVL;     // 33280 rows
    const int MQ  = Bsz * QL;      // 512 rows

    // K and V projections (concat(target, hidden) x W^T), N = 1024
    {
        int waves  = (MKV / 16) * (1024 / 64);   // 33280
        int blocks = waves / 8;
        proj_kernel<<<blocks, 256, 0, stream>>>(hidden, target, Wk, Kbuf, MKV, 1024, 1, 0);
        proj_kernel<<<blocks, 256, 0, stream>>>(hidden, target, Wv, Vt,   MKV, 1024, 1, 1);
    }
    // Q projection, N = 4096
    {
        int waves  = (MQ / 16) * (4096 / 64);    // 2048
        proj_kernel<<<waves / 8, 256, 0, stream>>>(hidden, nullptr, Wq, Qb, MQ, 4096, 0, 2);
    }
    // RMSNorm + RoPE (in place)
    {
        int krows = Bsz * NKVH * KVL;            // 266240
        norm_rope_kernel<<<(krows * 32) / 256, 256, 0, stream>>>(Kbuf, knw, cosb, sinb,
                                                                 krows, KVL, 0);
        int qrows = Bsz * NH * QL;               // 16384
        norm_rope_kernel<<<(qrows * 32) / 256, 256, 0, stream>>>(Qb, qnw, cosb, sinb,
                                                                 qrows, QL, CTX);
    }
    // attention: 1024 waves -> 128 blocks (8 waves/block share one (b,h))
    attn_kernel<<<128, 256, 0, stream>>>(Qb, Kbuf, Vt, Ob);
    // output projection -> fp32 d_out
    {
        int waves = (MQ / 16) * (4096 / 64);     // 2048
        oproj_kernel<<<waves / 8, 256, 0, stream>>>(Ob, Wo, (float*)d_out, MQ, 4096);
    }
}
